// HeteroGCNGNN_62079457296457
// MI455X (gfx1250) — matrix operand order repaired and checked by reading, not compiled, not run on hardware
//
#include <hip/hip_runtime.h>

typedef __bf16 bf16_t;
typedef __attribute__((ext_vector_type(16))) __bf16 v16bf;
typedef __attribute__((ext_vector_type(8)))  __bf16 v8bf;
typedef __attribute__((ext_vector_type(8)))  float  v8f;

#define NN     20000      // nodes per type
#define NE     600000     // edges per relation
#define DF     128        // feature dim
#define NL     2          // layers
#define NOUT   16
#define MTILES (NN / 16)  // 1250, exact

union AV { v16bf v; v8bf h[2]; };

// ---------------- elementwise / prep kernels ----------------

__global__ void k_f32_to_bf16(const float* __restrict__ x, bf16_t* __restrict__ y, int n) {
    int i = blockIdx.x * blockDim.x + threadIdx.x;
    if (i < n) y[i] = (bf16_t)x[i];
}

// W_conv [NL,4,DF,DF] (k-major rows) -> WT [NL,4,DF(n),DF(k)] bf16 (transposed)
__global__ void k_pack_wconv(const float* __restrict__ W, bf16_t* __restrict__ WT) {
    int i = blockIdx.x * blockDim.x + threadIdx.x;
    if (i >= NL * 4 * DF * DF) return;
    int mat = i >> 14;            // l*4+rel
    int r   = i & (DF * DF - 1);
    int n   = r >> 7;
    int k   = r & (DF - 1);
    WT[i] = (bf16_t)W[(size_t)mat * DF * DF + (size_t)k * DF + n];
}

// W_lin [DF,NOUT] x2 -> WTl [2,NOUT,DF] bf16
__global__ void k_pack_wlin(const float* __restrict__ Wg, const float* __restrict__ Wr,
                            bf16_t* __restrict__ WT) {
    int i = blockIdx.x * blockDim.x + threadIdx.x;
    if (i >= 2 * NOUT * DF) return;
    int t = i >> 11;              // NOUT*DF = 2048
    int r = i & 2047;
    int n = r >> 7;
    int k = r & 127;
    const float* W = t ? Wr : Wg;
    WT[i] = (bf16_t)W[(size_t)k * NOUT + n];
}

__global__ void k_zero(float* __restrict__ p, int n) {
    int i = blockIdx.x * blockDim.x + threadIdx.x;
    if (i < n) p[i] = 0.0f;
}

__global__ void k_deg(const int* __restrict__ idx, const float* __restrict__ w,
                      float* __restrict__ deg, int e_) {
    int e = blockIdx.x * blockDim.x + threadIdx.x;
    if (e < e_) atomicAdd(&deg[idx[e]], w[e]);
}

// deg -> dinv in place; regions [0,2*NN) are the self-loop relations (gg, rr): +1
__global__ void k_dinv(float* __restrict__ deg, int n) {
    int i = blockIdx.x * blockDim.x + threadIdx.x;
    if (i >= n) return;
    float d = deg[i] + ((i < 2 * NN) ? 1.0f : 0.0f);
    deg[i] = d > 0.0f ? rsqrtf(d) : 0.0f;
}

__global__ void k_norm(const int* __restrict__ ei, const float* __restrict__ w,
                       const float* __restrict__ dinv_s, const float* __restrict__ dinv_d,
                       float* __restrict__ norm) {
    int e = blockIdx.x * blockDim.x + threadIdx.x;
    if (e >= NE) return;
    norm[e] = dinv_s[ei[e]] * w[e] * dinv_d[ei[NE + e]];
}

// ---------------- WMMA GEMM: H[rel] = X(rel) @ W[rel]  (bf16 in, f32 out) ----

__global__ void __launch_bounds__(256)
k_gemm_h(const bf16_t* __restrict__ Xg, const bf16_t* __restrict__ Xr,
         const bf16_t* __restrict__ WT,   // [4,DF(n),DF(k)] layer slice, pre-transposed
         float* __restrict__ H) {         // [4,NN,DF]
    int rel = blockIdx.y;
    const bf16_t* X = (rel < 2) ? Xg : Xr;
    const bf16_t* W = WT + (size_t)rel * DF * DF;
    float* Ho = H + (size_t)rel * NN * DF;

    int mtile = blockIdx.x;             // 0..1249
    int wave  = threadIdx.x >> 5;       // 0..7 -> N-tile
    int lane  = threadIdx.x & 31;
    int l16   = lane & 15;
    int hi    = lane >> 4;              // K-half selector

    const bf16_t* xrow = X + (size_t)(mtile * 16 + l16) * DF;   // A row (M = l16)
    const bf16_t* wrow = W + (size_t)(wave * 16 + l16) * DF;    // B col (N = l16) as row of WT

    v8f c = {};
#pragma unroll
    for (int k0 = 0; k0 < DF; k0 += 32) {
        AV a;
        a.h[0] = *(const v8bf*)(xrow + k0 + hi * 8);        // K 0-7 / 8-15
        a.h[1] = *(const v8bf*)(xrow + k0 + hi * 8 + 16);   // K 16-23 / 24-31
        AV bv;
        bv.h[0] = *(const v8bf*)(wrow + k0 + hi * 8);       // K 0-7 / 8-15
        bv.h[1] = *(const v8bf*)(wrow + k0 + hi * 8 + 16);  // K 16-23 / 24-31
        v16bf b = bv.v;
        c = __builtin_amdgcn_wmma_f32_16x16x32_bf16(false, a.v, false, b,
                                                    (short)0, c, false, false);
    }
    float* orow = Ho + (size_t)mtile * 16 * DF + wave * 16 + l16;
#pragma unroll
    for (int j = 0; j < 8; j++)
        orow[(size_t)(j + hi * 8) * DF] = c[j];
}

// ---------------- aggregation ----------------

// self-loop contribution: agg_gen = h_gg * dinv_gg^2 ; agg_rain = h_rr * dinv_rr^2
__global__ void k_agg_init(const float* __restrict__ H, const float* __restrict__ dinv,
                           float* __restrict__ agg_gen, float* __restrict__ agg_rain) {
    int i = blockIdx.x * blockDim.x + threadIdx.x;
    if (i >= NN * DF) return;
    int node = i >> 7;
    float dg = dinv[node];           // dinv_gg region
    float dr = dinv[NN + node];      // dinv_rr region
    agg_gen[i]  = H[i] * dg * dg;                       // slab 0 = h_gg
    agg_rain[i] = H[(size_t)3 * NN * DF + i] * dr * dr; // slab 3 = h_rr
}

// one wave per edge; lane owns 4 contiguous features (float4 gather, 4 f32 atomics)
__global__ void __launch_bounds__(256)
k_scatter(const float* __restrict__ H, const int* __restrict__ ei,
          const float* __restrict__ norm, float* __restrict__ agg) {
    int e = blockIdx.x * (blockDim.x >> 5) + (threadIdx.x >> 5);
    if (e >= NE) return;
    int lane = threadIdx.x & 31;
    int src = ei[e];
    int dst = ei[NE + e];
    float nrm = norm[e];
    const float4 h4 = *(const float4*)(H + (size_t)src * DF + lane * 4);
    float* d = agg + (size_t)dst * DF + lane * 4;
    atomicAdd(d + 0, h4.x * nrm);
    atomicAdd(d + 1, h4.y * nrm);
    atomicAdd(d + 2, h4.z * nrm);
    atomicAdd(d + 3, h4.w * nrm);
}

// mean over 2 relations + biases + ReLU, re-quantize to bf16 for next WMMA layer
__global__ void k_combine(const float* __restrict__ agg_gen, const float* __restrict__ agg_rain,
                          const float* __restrict__ bl,   // b_conv layer slice [4,DF]
                          bf16_t* __restrict__ xg, bf16_t* __restrict__ xr) {
    int i = blockIdx.x * blockDim.x + threadIdx.x;
    if (i >= NN * DF) return;
    int f = i & 127;
    float vg = (agg_gen[i]  + bl[f]      + bl[2 * DF + f]) * 0.5f;
    float vr = (agg_rain[i] + bl[DF + f] + bl[3 * DF + f]) * 0.5f;
    xg[i] = (bf16_t)fmaxf(vg, 0.0f);
    xr[i] = (bf16_t)fmaxf(vr, 0.0f);
}

// ---------------- output heads: out = X @ W_lin + b (WMMA, N-tile = 16 = OUT) --

__global__ void __launch_bounds__(256)
k_lin(const bf16_t* __restrict__ Xg, const bf16_t* __restrict__ Xr,
      const bf16_t* __restrict__ WT,  // [2,NOUT,DF]
      const float* __restrict__ bg, const float* __restrict__ br,
      float* __restrict__ out) {      // gen[NN,16] then rain[NN,16]
    int type = blockIdx.y;
    const bf16_t* X = type ? Xr : Xg;
    const bf16_t* W = WT + (size_t)type * NOUT * DF;
    const float* bias = type ? br : bg;
    float* o = out + (size_t)type * NN * NOUT;

    int mtile = blockIdx.x * (blockDim.x >> 5) + (threadIdx.x >> 5);
    if (mtile >= MTILES) return;      // uniform per wave -> EXEC stays full
    int lane = threadIdx.x & 31;
    int l16  = lane & 15;
    int hi   = lane >> 4;

    const bf16_t* xrow = X + (size_t)(mtile * 16 + l16) * DF;
    const bf16_t* wrow = W + (size_t)l16 * DF;

    v8f c = {};
#pragma unroll
    for (int k0 = 0; k0 < DF; k0 += 32) {
        AV a;
        a.h[0] = *(const v8bf*)(xrow + k0 + hi * 8);
        a.h[1] = *(const v8bf*)(xrow + k0 + hi * 8 + 16);
        AV bv;
        bv.h[0] = *(const v8bf*)(wrow + k0 + hi * 8);
        bv.h[1] = *(const v8bf*)(wrow + k0 + hi * 8 + 16);
        v16bf b = bv.v;
        c = __builtin_amdgcn_wmma_f32_16x16x32_bf16(false, a.v, false, b,
                                                    (short)0, c, false, false);
    }
    float bb = bias[l16];
    float* orow = o + (size_t)mtile * 16 * NOUT + l16;
#pragma unroll
    for (int j = 0; j < 8; j++)
        orow[(size_t)(j + hi * 8) * NOUT] = c[j] + bb;
}

// ---------------- host driver ----------------

extern "C" void kernel_launch(void* const* d_in, const int* in_sizes, int n_in,
                              void* d_out, int out_size, void* d_ws, size_t ws_size,
                              hipStream_t stream) {
    (void)in_sizes; (void)n_in; (void)out_size; (void)ws_size;

    const float* x_gen  = (const float*)d_in[0];
    const float* x_rain = (const float*)d_in[1];
    const int*   ei_gg  = (const int*)d_in[2];
    const int*   ei_gr  = (const int*)d_in[3];
    const int*   ei_rg  = (const int*)d_in[4];
    const int*   ei_rr  = (const int*)d_in[5];
    const float* w_gg   = (const float*)d_in[6];
    const float* w_gr   = (const float*)d_in[7];
    const float* w_rg   = (const float*)d_in[8];
    const float* w_rr   = (const float*)d_in[9];
    const float* W_conv = (const float*)d_in[10];
    const float* b_conv = (const float*)d_in[11];
    const float* W_lg   = (const float*)d_in[12];
    const float* b_lg   = (const float*)d_in[13];
    const float* W_lr   = (const float*)d_in[14];
    const float* b_lr   = (const float*)d_in[15];
    float* out = (float*)d_out;

    char* ws = (char*)d_ws;
    size_t off = 0;
    auto alloc = [&](size_t bytes) -> void* {
        void* p = ws + off;
        off += (bytes + 255) & ~(size_t)255;
        return p;
    };
    bf16_t* xg_b  = (bf16_t*)alloc((size_t)NN * DF * 2);
    bf16_t* xr_b  = (bf16_t*)alloc((size_t)NN * DF * 2);
    bf16_t* WT    = (bf16_t*)alloc((size_t)NL * 4 * DF * DF * 2);
    bf16_t* WTl   = (bf16_t*)alloc((size_t)2 * NOUT * DF * 2);
    float*  H     = (float*)alloc((size_t)4 * NN * DF * 4);
    float*  agg_g = (float*)alloc((size_t)NN * DF * 4);
    float*  agg_r = (float*)alloc((size_t)NN * DF * 4);
    float*  dinv  = (float*)alloc((size_t)6 * NN * 4);   // gg, rr, gr_s, gr_d, rg_s, rg_d
    float*  n_gg  = (float*)alloc((size_t)NE * 4);
    float*  n_gr  = (float*)alloc((size_t)NE * 4);
    float*  n_rg  = (float*)alloc((size_t)NE * 4);
    float*  n_rr  = (float*)alloc((size_t)NE * 4);

    const int TB = 256;
    const int EB = (NE + TB - 1) / TB;

    // prep: bf16 activations + pre-transposed bf16 weights
    k_f32_to_bf16<<<(NN * DF + TB - 1) / TB, TB, 0, stream>>>(x_gen,  xg_b, NN * DF);
    k_f32_to_bf16<<<(NN * DF + TB - 1) / TB, TB, 0, stream>>>(x_rain, xr_b, NN * DF);
    k_pack_wconv<<<(NL * 4 * DF * DF + TB - 1) / TB, TB, 0, stream>>>(W_conv, WT);
    k_pack_wlin<<<(2 * NOUT * DF + TB - 1) / TB, TB, 0, stream>>>(W_lg, W_lr, WTl);

    // degrees -> dinv -> per-edge norms (layer-invariant)
    k_zero<<<(6 * NN + TB - 1) / TB, TB, 0, stream>>>(dinv, 6 * NN);
    k_deg<<<EB, TB, 0, stream>>>(ei_gg + NE, w_gg, dinv + 0 * NN, NE);  // gg dst
    k_deg<<<EB, TB, 0, stream>>>(ei_rr + NE, w_rr, dinv + 1 * NN, NE);  // rr dst
    k_deg<<<EB, TB, 0, stream>>>(ei_gr,      w_gr, dinv + 2 * NN, NE);  // gr src (gen)
    k_deg<<<EB, TB, 0, stream>>>(ei_gr + NE, w_gr, dinv + 3 * NN, NE);  // gr dst (rain)
    k_deg<<<EB, TB, 0, stream>>>(ei_rg,      w_rg, dinv + 4 * NN, NE);  // rg src (rain)
    k_deg<<<EB, TB, 0, stream>>>(ei_rg + NE, w_rg, dinv + 5 * NN, NE);  // rg dst (gen)
    k_dinv<<<(6 * NN + TB - 1) / TB, TB, 0, stream>>>(dinv, 6 * NN);
    k_norm<<<EB, TB, 0, stream>>>(ei_gg, w_gg, dinv + 0 * NN, dinv + 0 * NN, n_gg);
    k_norm<<<EB, TB, 0, stream>>>(ei_gr, w_gr, dinv + 2 * NN, dinv + 3 * NN, n_gr);
    k_norm<<<EB, TB, 0, stream>>>(ei_rg, w_rg, dinv + 4 * NN, dinv + 5 * NN, n_rg);
    k_norm<<<EB, TB, 0, stream>>>(ei_rr, w_rr, dinv + 1 * NN, dinv + 1 * NN, n_rr);

    const int SB = NE / 8;  // one wave per edge, 8 waves per block (600000 % 8 == 0)
    for (int l = 0; l < NL; l++) {
        k_gemm_h<<<dim3(MTILES, 4), 256, 0, stream>>>(xg_b, xr_b,
                                                      WT + (size_t)l * 4 * DF * DF, H);
        k_agg_init<<<(NN * DF + TB - 1) / TB, TB, 0, stream>>>(H, dinv, agg_g, agg_r);
        k_scatter<<<SB, 256, 0, stream>>>(H + (size_t)0 * NN * DF, ei_gg, n_gg, agg_g);
        k_scatter<<<SB, 256, 0, stream>>>(H + (size_t)1 * NN * DF, ei_gr, n_gr, agg_r);
        k_scatter<<<SB, 256, 0, stream>>>(H + (size_t)2 * NN * DF, ei_rg, n_rg, agg_g);
        k_scatter<<<SB, 256, 0, stream>>>(H + (size_t)3 * NN * DF, ei_rr, n_rr, agg_r);
        k_combine<<<(NN * DF + TB - 1) / TB, TB, 0, stream>>>(agg_g, agg_r,
                                                              b_conv + (size_t)l * 4 * DF,
                                                              xg_b, xr_b);
    }

    k_lin<<<dim3((MTILES + 7) / 8, 2), 256, 0, stream>>>(xg_b, xr_b, WTl, b_lg, b_lr, out);
}
